// ProductionHybridModel_33809982554131
// MI455X (gfx1250) — matrix operand order repaired and checked by reading, not compiled, not run on hardware
//
#include <hip/hip_runtime.h>
#include <math.h>

// ---------------------------------------------------------------------------
// Types for CDNA5 WMMA (wave32, 16x16x32 f16 -> f32)
// ---------------------------------------------------------------------------
typedef _Float16 h16;
typedef __attribute__((ext_vector_type(16))) _Float16 v16h;
typedef __attribute__((ext_vector_type(8)))  _Float16 h8;
typedef __attribute__((ext_vector_type(8)))  float    v8f;

#define B_   4
#define L_   1024
#define D_   1024
#define DI_  2048
#define DSTATE 16
#define NH   16
#define NKV  4
#define DH   64
#define DFF  4096
#define T_   (B_ * L_)

__device__ __forceinline__ float silu_f(float x)     { return x / (1.f + __expf(-x)); }
__device__ __forceinline__ float softplus_f(float x) { return (x > 20.f) ? x : log1pf(__expf(x)); }

__device__ __forceinline__ v16h cat8(h8 lo, h8 hi) {
  return __builtin_shufflevector(lo, hi, 0,1,2,3,4,5,6,7,8,9,10,11,12,13,14,15);
}

// ---------------------------------------------------------------------------
// Generic WMMA GEMM: C[M,N] = act(A[M,K]@W[K,N] + bias) + addsrc
// A,W row-major f16; C f32; optional f16 mirror Ch. Block tile 64x128, BK=32.
// 8 waves (2x4), each wave 32x32 via 2x2 v_wmma_f32_16x16x32_f16.
// ACT: 0=none 1=silu 2=softplus
// ---------------------------------------------------------------------------
template <int ACT>
__global__ void __launch_bounds__(256)
gemm_f16_wmma(const h16* __restrict__ A, int lda,
              const h16* __restrict__ W, int ldw,
              float* __restrict__ C, int ldc,
              h16* __restrict__ Ch,
              const float* __restrict__ bias,
              const float* __restrict__ addsrc,
              int M, int N, int K)
{
  __shared__ __align__(16) h16 As[64][40];    // 64 rows x 32 K (+pad)
  __shared__ __align__(16) h16 Bs[32][136];   // 32 K x 128 N (+pad)

  const int tid  = threadIdx.x;
  const int lane = tid & 31, l16 = lane & 15, hi = lane >> 4;
  const int wave = tid >> 5;
  const int wm = wave >> 2, wn = wave & 3;    // waves 2 (M) x 4 (N)
  const int m0 = blockIdx.y * 64;
  const int n0 = blockIdx.x * 128;
  const h16 hz = (h16)0.f;

  v8f acc[2][2];
  #pragma unroll
  for (int a = 0; a < 2; ++a)
    #pragma unroll
    for (int b = 0; b < 2; ++b) acc[a][b] = {};

  const int nk = (K + 31) >> 5;
  for (int kt = 0; kt < nk; ++kt) {
    const int k0 = kt << 5;
    // ---- stage A tile (64x32): each thread loads 8 f16 -------------------
    {
      const int r = tid >> 2;
      const int c = (tid & 3) << 3;
      const int gr = m0 + r;
      #pragma unroll
      for (int e = 0; e < 8; ++e) {
        const int gk = k0 + c + e;
        As[r][c + e] = (gr < M && gk < K) ? A[(size_t)gr * lda + gk] : hz;
      }
      if (gr < M && k0 + 32 < K)
        __builtin_prefetch(&A[(size_t)gr * lda + k0 + 32], 0, 0);  // global_prefetch_b8
    }
    // ---- stage W tile (32x128) row-major ---------------------------------
    {
      #pragma unroll
      for (int e = 0; e < 16; ++e) {
        const int idx = e * 256 + tid;      // 0..4095
        const int k = idx >> 7;             // 0..31
        const int n = idx & 127;            // 0..127
        const int gk = k0 + k, gn = n0 + n;
        Bs[k][n] = (gk < K && gn < N) ? W[(size_t)gk * ldw + gn] : hz;
      }
      if (k0 + 32 < K)
        __builtin_prefetch(&W[(size_t)(k0 + 32) * ldw + n0 + (tid & 127)], 0, 0);
    }
    __syncthreads();

    // ---- fragments -------------------------------------------------------
    // A frag: row per lane; lanes 0-15 hold K 0-7 & 16-23, lanes 16-31 hold 8-15 & 24-31
    v16h af[2], bf[2];
    #pragma unroll
    for (int ms = 0; ms < 2; ++ms) {
      const int r = wm * 32 + ms * 16 + l16;
      const int base = hi * 8;
      h8 lo = *(const h8*)&As[r][base];
      h8 hh = *(const h8*)&As[r][base + 16];
      af[ms] = cat8(lo, hh);
    }
    // B frag: column N per lane; lanes 0-15 hold K 0-7 & 16-23, lanes 16-31 hold 8-15 & 24-31
    #pragma unroll
    for (int ns = 0; ns < 2; ++ns) {
      const int cb = wn * 32 + ns * 16 + l16;
      #pragma unroll
      for (int s = 0; s < 16; ++s)
        bf[ns][s] = Bs[hi * 8 + (s & 7) + ((s >> 3) << 4)][cb];
    }
    #pragma unroll
    for (int ms = 0; ms < 2; ++ms)
      #pragma unroll
      for (int ns = 0; ns < 2; ++ns)
        acc[ms][ns] = __builtin_amdgcn_wmma_f32_16x16x32_f16(
            false, af[ms], false, bf[ns], (short)0, acc[ms][ns], false, false);
    __syncthreads();
  }

  // ---- epilogue: C layout = VGPR i, lane -> (M = i + 8*hi, N = l16) -------
  #pragma unroll
  for (int ms = 0; ms < 2; ++ms)
    #pragma unroll
    for (int ns = 0; ns < 2; ++ns)
      #pragma unroll
      for (int i = 0; i < 8; ++i) {
        const int row = m0 + wm * 32 + ms * 16 + i + hi * 8;
        const int col = n0 + wn * 32 + ns * 16 + l16;
        if (row < M && col < N) {
          float v = acc[ms][ns][i];
          if (bias) v += bias[col];
          if (ACT == 1) v = silu_f(v);
          else if (ACT == 2) v = softplus_f(v);
          if (addsrc) v += addsrc[(size_t)row * ldc + col];
          C[(size_t)row * ldc + col] = v;
          if (Ch) Ch[(size_t)row * ldc + col] = (h16)v;
        }
      }
}

// ---------------------------------------------------------------------------
// LayerNorm over D=1024, one block per token, f16 output
// ---------------------------------------------------------------------------
__global__ void __launch_bounds__(256)
ln_kernel(const float* __restrict__ x, const float* __restrict__ g,
          const float* __restrict__ bb, h16* __restrict__ out)
{
  __shared__ float red[256];
  const int row = blockIdx.x;
  const float* xr = x + (size_t)row * D_;
  float s = 0.f, ss = 0.f;
  for (int i = threadIdx.x; i < D_; i += 256) { float v = xr[i]; s += v; ss += v * v; }
  red[threadIdx.x] = s; __syncthreads();
  for (int st = 128; st > 0; st >>= 1) { if (threadIdx.x < st) red[threadIdx.x] += red[threadIdx.x + st]; __syncthreads(); }
  const float mean = red[0] * (1.f / D_);
  __syncthreads();
  red[threadIdx.x] = ss; __syncthreads();
  for (int st = 128; st > 0; st >>= 1) { if (threadIdx.x < st) red[threadIdx.x] += red[threadIdx.x + st]; __syncthreads(); }
  const float var = red[0] * (1.f / D_) - mean * mean;
  const float rstd = rsqrtf(var + 1e-5f);
  for (int i = threadIdx.x; i < D_; i += 256)
    out[(size_t)row * D_ + i] = (h16)((xr[i] - mean) * rstd * g[i] + bb[i]);
}

// ---------------------------------------------------------------------------
// f32 -> f16 convert
// ---------------------------------------------------------------------------
__global__ void cvt_kernel(const float* __restrict__ s, h16* __restrict__ d, int n) {
  const int i = blockIdx.x * 256 + threadIdx.x;
  if (i < n) d[i] = (h16)s[i];
}

// ---------------------------------------------------------------------------
// Causal depthwise conv (d_conv=4) + bias + SiLU. xs = xs_gate[:, :2048]
// ---------------------------------------------------------------------------
__global__ void __launch_bounds__(256)
conv_silu_kernel(const float* __restrict__ xs_gate, const float* __restrict__ cw,
                 const float* __restrict__ cb, float* __restrict__ xsc,
                 h16* __restrict__ xsch)
{
  const int idx = blockIdx.x * 256 + threadIdx.x;
  if (idx >= T_ * DI_) return;
  const int c = idx & (DI_ - 1);
  const int l = (idx >> 11) & (L_ - 1);
  const int b = idx >> 21;
  float acc = cb[c];
  #pragma unroll
  for (int j = 0; j < 4; ++j) {
    const int t = l - 3 + j;
    if (t >= 0) acc += xs_gate[((size_t)(b * L_ + t)) * (2 * DI_) + c] * cw[c * 4 + j];
  }
  const float v = silu_f(acc);
  xsc[idx] = v;
  xsch[idx] = (h16)v;
}

// ---------------------------------------------------------------------------
// Sequential selective scan: one lane per (batch, channel), 16 states in regs
// ---------------------------------------------------------------------------
__global__ void __launch_bounds__(256)
scan_kernel(const float* __restrict__ xsc, const float* __restrict__ dt,
            const float* __restrict__ BC, const float* __restrict__ A_log,
            float* __restrict__ y)
{
  const int gid = blockIdx.x * 256 + threadIdx.x;   // 0..8191
  const int c = gid & (DI_ - 1);
  const int b = gid >> 11;
  float Ar[DSTATE], hs[DSTATE];
  #pragma unroll
  for (int s = 0; s < DSTATE; ++s) { Ar[s] = -__expf(A_log[c * DSTATE + s]); hs[s] = 0.f; }
  for (int t = 0; t < L_; ++t) {
    const size_t rowi = (size_t)(b * L_ + t);
    const float x = xsc[rowi * DI_ + c];
    const float d = dt[rowi * DI_ + c];
    const float* bc = BC + rowi * 32;
    float yt = 0.f;
    #pragma unroll
    for (int s = 0; s < DSTATE; ++s) {
      hs[s] = hs[s] * __expf(d * Ar[s]) + d * bc[s] * x;
      yt += hs[s] * bc[16 + s];
    }
    y[rowi * DI_ + c] = yt;
  }
}

// ---------------------------------------------------------------------------
// Mamba gate: yg = (y + D*xs) * silu(gate)    (f16 out for next GEMM)
// ---------------------------------------------------------------------------
__global__ void __launch_bounds__(256)
gate_kernel(const float* __restrict__ y, const float* __restrict__ xsc,
            const float* __restrict__ xs_gate, const float* __restrict__ Dskip,
            h16* __restrict__ ygh)
{
  const int idx = blockIdx.x * 256 + threadIdx.x;
  if (idx >= T_ * DI_) return;
  const int c = idx & (DI_ - 1);
  const int row = idx >> 11;
  const float g = xs_gate[(size_t)row * (2 * DI_) + DI_ + c];
  ygh[idx] = (h16)((y[idx] + Dskip[c] * xsc[idx]) * silu_f(g));
}

// ---------------------------------------------------------------------------
// RoPE for Q: writes [B][H][L][64] f16
// ---------------------------------------------------------------------------
__global__ void __launch_bounds__(256)
rope_q_kernel(const float* __restrict__ q, h16* __restrict__ qh)
{
  const int idx = blockIdx.x * 256 + threadIdx.x;
  if (idx >= B_ * NH * L_ * 32) return;
  const int j = idx & 31;
  const int l = (idx >> 5) & (L_ - 1);
  const int h = (idx >> 15) & (NH - 1);
  const int b = idx >> 19;
  const float inv = __powf(10000.f, -(float)(2 * j) * (1.f / 64.f));
  const float a = (float)l * inv;
  const float c = __cosf(a), s = __sinf(a);
  const float* src = q + ((size_t)(b * L_ + l)) * (NH * DH) + h * DH;
  const float x1 = src[j], x2 = src[j + 32];
  const size_t dst = ((size_t)(b * NH + h) * L_ + l) * DH;
  qh[dst + j]      = (h16)(x1 * c - x2 * s);
  qh[dst + j + 32] = (h16)(x2 * c + x1 * s);
}

// RoPE for K: writes TRANSPOSED [B][KV][64][L] f16 (contiguous keys per d)
__global__ void __launch_bounds__(256)
rope_k_kernel(const float* __restrict__ k, h16* __restrict__ kt)
{
  const int idx = blockIdx.x * 256 + threadIdx.x;
  if (idx >= B_ * NKV * L_ * 32) return;
  const int j  = idx & 31;
  const int l  = (idx >> 5) & (L_ - 1);
  const int kv = (idx >> 15) & (NKV - 1);
  const int b  = idx >> 17;
  const float inv = __powf(10000.f, -(float)(2 * j) * (1.f / 64.f));
  const float a = (float)l * inv;
  const float c = __cosf(a), s = __sinf(a);
  const float* src = k + ((size_t)(b * L_ + l)) * (NKV * DH) + kv * DH;
  const float x1 = src[j], x2 = src[j + 32];
  const size_t db = (size_t)(b * NKV + kv) * DH;
  kt[(db + j) * L_ + l]      = (h16)(x1 * c - x2 * s);
  kt[(db + j + 32) * L_ + l] = (h16)(x2 * c + x1 * s);
}

// V: reshape to [B][KV][L][64] f16
__global__ void __launch_bounds__(256)
vtrans_kernel(const float* __restrict__ v, h16* __restrict__ vh)
{
  const int idx = blockIdx.x * 256 + threadIdx.x;
  if (idx >= B_ * L_ * NKV * DH) return;
  const int d  = idx & 63;
  const int kv = (idx >> 6) & 3;
  const int l  = (idx >> 8) & (L_ - 1);
  const int b  = idx >> 18;
  vh[(((size_t)(b * NKV + kv)) * L_ + l) * DH + d] =
      (h16)v[((size_t)(b * L_ + l)) * (NKV * DH) + kv * DH + d];
}

// ---------------------------------------------------------------------------
// Flash attention: 1 wave per block, 16 queries/block, key tiles of 32,
// causal mask + online softmax, all matmuls via WMMA. GQA: head h -> kv h/4.
// ---------------------------------------------------------------------------
__global__ void __launch_bounds__(32)
flash_attn(const h16* __restrict__ qh, const h16* __restrict__ kt_,
           const h16* __restrict__ vh, h16* __restrict__ oh)
{
  __shared__ __align__(16) h16 Pl[16][32];
  const int lane = threadIdx.x & 31, l16 = lane & 15, hi = lane >> 4;
  const int q0 = blockIdx.x * 16;
  const int h  = blockIdx.y;
  const int b  = blockIdx.z;
  const int kv = h >> 2;
  const float scale = 0.125f;   // 1/sqrt(64)

  // Q A-fragments (two 32-wide d chunks)
  v16h aq[2];
  {
    const h16* qp = qh + (((size_t)(b * NH + h)) * L_ + q0 + l16) * DH;
    #pragma unroll
    for (int ks = 0; ks < 2; ++ks) {
      const int base = ks * 32 + hi * 8;
      aq[ks] = cat8(*(const h8*)&qp[base], *(const h8*)&qp[base + 16]);
    }
  }
  v8f o[4];
  #pragma unroll
  for (int n = 0; n < 4; ++n) o[n] = {};
  float mst[8], lst[8];
  #pragma unroll
  for (int i = 0; i < 8; ++i) { mst[i] = -1e30f; lst[i] = 0.f; }

  const int tkmax = (q0 + 15) >> 5;
  for (int tk = 0; tk <= tkmax; ++tk) {
    // ---- scores S = Q @ K^T  (16 x 32) -----------------------------------
    float pv[2][8];
    #pragma unroll
    for (int ns = 0; ns < 2; ++ns) {
      v8f s = {};
      #pragma unroll
      for (int ks = 0; ks < 2; ++ks) {
        // B frag from transposed K: key column per lane, d = ks*32 + hi*8 + {0-7, 16-23}
        const h16* kp = kt_ + ((((size_t)(b * NKV + kv)) * DH + ks * 32 + hi * 8) * L_)
                        + tk * 32 + ns * 16 + l16;
        v16h bk;
        #pragma unroll
        for (int s = 0; s < 16; ++s) bk[s] = kp[(size_t)((s & 7) + ((s >> 3) << 4)) * L_];
        s = __builtin_amdgcn_wmma_f32_16x16x32_f16(false, aq[ks], false, bk, (short)0, s, false, false);
      }
      #pragma unroll
      for (int i = 0; i < 8; ++i) {
        const int row = q0 + i + hi * 8;
        const int col = tk * 32 + ns * 16 + l16;
        float sv = s[i] * scale;
        if (col > row) sv = -1e30f;
        pv[ns][i] = sv;
      }
    }
    // ---- online softmax (row = i + 8*hi lives in one 16-lane half) -------
    #pragma unroll
    for (int i = 0; i < 8; ++i) {
      float mx = fmaxf(pv[0][i], pv[1][i]);
      mx = fmaxf(mx, __shfl_xor(mx, 1, 32));
      mx = fmaxf(mx, __shfl_xor(mx, 2, 32));
      mx = fmaxf(mx, __shfl_xor(mx, 4, 32));
      mx = fmaxf(mx, __shfl_xor(mx, 8, 32));
      const float mnew = fmaxf(mst[i], mx);
      const float corr = __expf(mst[i] - mnew);
      mst[i] = mnew;
      const float p0 = __expf(pv[0][i] - mnew);
      const float p1 = __expf(pv[1][i] - mnew);
      pv[0][i] = p0; pv[1][i] = p1;
      float ls = p0 + p1;
      ls += __shfl_xor(ls, 1, 32);
      ls += __shfl_xor(ls, 2, 32);
      ls += __shfl_xor(ls, 4, 32);
      ls += __shfl_xor(ls, 8, 32);
      lst[i] = lst[i] * corr + ls;
      #pragma unroll
      for (int n = 0; n < 4; ++n) o[n][i] *= corr;
    }
    // ---- P through LDS: C layout -> A layout -----------------------------
    __syncthreads();
    #pragma unroll
    for (int ns = 0; ns < 2; ++ns)
      #pragma unroll
      for (int i = 0; i < 8; ++i)
        Pl[i + hi * 8][ns * 16 + l16] = (h16)pv[ns][i];
    __syncthreads();
    v16h ap;
    {
      const int base = hi * 8;
      ap = cat8(*(const h8*)&Pl[l16][base], *(const h8*)&Pl[l16][base + 16]);
    }
    // ---- O += P @ V ------------------------------------------------------
    #pragma unroll
    for (int n = 0; n < 4; ++n) {
      const h16* vp = vh + (((size_t)(b * NKV + kv)) * L_ + tk * 32 + hi * 8) * DH + n * 16 + l16;
      v16h bv;
      #pragma unroll
      for (int s = 0; s < 16; ++s) bv[s] = vp[((s & 7) + ((s >> 3) << 4)) * DH];
      o[n] = __builtin_amdgcn_wmma_f32_16x16x32_f16(false, ap, false, bv, (short)0, o[n], false, false);
    }
  }
  // ---- write O / l  ->  [B][L][H*64] f16 ---------------------------------
  #pragma unroll
  for (int n = 0; n < 4; ++n)
    #pragma unroll
    for (int i = 0; i < 8; ++i) {
      const int row = q0 + i + hi * 8;
      const int d = n * 16 + l16;
      oh[((size_t)(b * L_ + row)) * (NH * DH) + h * DH + d] = (h16)(o[n][i] / lst[i]);
    }
}

// ---------------------------------------------------------------------------
// SwiGLU combine: gh = (f16)(silu(u1) * u3); u1 already silu'd in GEMM epilogue
// ---------------------------------------------------------------------------
__global__ void __launch_bounds__(256)
swiglu_kernel(const float* __restrict__ u1, const float* __restrict__ u3,
              h16* __restrict__ gh, int n)
{
  const int i = blockIdx.x * 256 + threadIdx.x;
  if (i < n) gh[i] = (h16)(u1[i] * u3[i]);
}

// ---------------------------------------------------------------------------
// Host-side orchestration
// ---------------------------------------------------------------------------
extern "C" void kernel_launch(void* const* d_in, const int* in_sizes, int n_in,
                              void* d_out, int out_size, void* d_ws, size_t ws_size,
                              hipStream_t stream) {
  (void)in_sizes; (void)n_in; (void)out_size; (void)ws_size;
  const float* x_in   = (const float*)d_in[0];
  const float* ln1_g  = (const float*)d_in[1];
  const float* ln1_b  = (const float*)d_in[2];
  const float* in_w   = (const float*)d_in[3];
  const float* conv_w = (const float*)d_in[4];
  const float* conv_b = (const float*)d_in[5];
  const float* xp_w   = (const float*)d_in[6];
  const float* dt_w   = (const float*)d_in[7];
  const float* dt_b   = (const float*)d_in[8];
  const float* A_log  = (const float*)d_in[9];
  const float* Dskip  = (const float*)d_in[10];
  const float* out_w  = (const float*)d_in[11];
  const float* ln2_g  = (const float*)d_in[12];
  const float* ln2_b  = (const float*)d_in[13];
  const float* q_w    = (const float*)d_in[14];
  const float* k_w    = (const float*)d_in[15];
  const float* v_w    = (const float*)d_in[16];
  const float* o_w    = (const float*)d_in[17];
  const float* ln3_g  = (const float*)d_in[18];
  const float* ln3_b  = (const float*)d_in[19];
  const float* w1     = (const float*)d_in[20];
  const float* w2     = (const float*)d_in[21];
  const float* w3     = (const float*)d_in[22];
  float* out = (float*)d_out;

  char* base = (char*)d_ws;
  size_t off = 0;
  auto alloc = [&](size_t bytes) -> void* {
    void* r = base + off;
    off += (bytes + 255) & ~(size_t)255;
    return r;
  };

  // f16 weights
  h16* in_w_h  = (h16*)alloc((size_t)D_ * 2 * DI_ * 2);
  h16* xp_h    = (h16*)alloc((size_t)DI_ * 32 * 2);
  h16* dt_h    = (h16*)alloc((size_t)DSTATE * DI_ * 2);
  h16* outw_h  = (h16*)alloc((size_t)DI_ * D_ * 2);
  h16* qw_h    = (h16*)alloc((size_t)D_ * D_ * 2);
  h16* kw_h    = (h16*)alloc((size_t)D_ * 256 * 2);
  h16* vw_h    = (h16*)alloc((size_t)D_ * 256 * 2);
  h16* ow_h    = (h16*)alloc((size_t)D_ * D_ * 2);
  h16* w1_h    = (h16*)alloc((size_t)D_ * DFF * 2);
  h16* w2_h    = (h16*)alloc((size_t)DFF * D_ * 2);
  h16* w3_h    = (h16*)alloc((size_t)D_ * DFF * 2);
  // activations
  h16*   h1_h    = (h16*)alloc((size_t)T_ * D_ * 2);
  float* xs_gate = (float*)alloc((size_t)T_ * 2 * DI_ * 4);   // 64 MB, reused as u1
  float* xsc     = (float*)alloc((size_t)T_ * DI_ * 4);
  h16*   xsc_h   = (h16*)alloc((size_t)T_ * DI_ * 2);
  float* BC      = (float*)alloc((size_t)T_ * 32 * 4);
  h16*   BC_h    = (h16*)alloc((size_t)T_ * 32 * 2);
  float* dtb     = (float*)alloc((size_t)T_ * DI_ * 4);       // 32 MB, contiguous with yb
  float* yb      = (float*)alloc((size_t)T_ * DI_ * 4);       // 32 MB, dtb+yb reused as u3
  h16*   yg_h    = (h16*)alloc((size_t)T_ * DI_ * 2);
  float* x1      = (float*)alloc((size_t)T_ * D_ * 4);
  h16*   h2_h    = (h16*)alloc((size_t)T_ * D_ * 2);
  float* qf      = (float*)alloc((size_t)T_ * D_ * 4);        // 16 MB, start of 32 MB run
  float* kf      = (float*)alloc((size_t)T_ * 256 * 4);       //  4 MB
  float* vf      = (float*)alloc((size_t)T_ * 256 * 4);       //  4 MB
  h16*   q_h     = (h16*)alloc((size_t)T_ * D_ * 2);          //  8 MB, qf..q_h reused as g_h
  h16*   k_t     = (h16*)alloc((size_t)T_ * 256 * 2);
  h16*   v_h     = (h16*)alloc((size_t)T_ * 256 * 2);
  h16*   o_h     = (h16*)alloc((size_t)T_ * D_ * 2);
  float* x2      = (float*)alloc((size_t)T_ * D_ * 4);
  h16*   h3_h    = (h16*)alloc((size_t)T_ * D_ * 2);
  // buffer reuse for FFN (earlier buffers are dead by then)
  float* u1 = xs_gate;            // 64 MB
  float* u3 = dtb;                // dtb+yb contiguous 64 MB
  h16*   g_h = (h16*)qf;          // qf..q_h contiguous 32 MB

  auto cvt = [&](const float* s, h16* d, int n) {
    cvt_kernel<<<dim3((n + 255) / 256), 256, 0, stream>>>(s, d, n);
  };
  auto gemm = [&](int act, const h16* A, int lda, const h16* W, int ldw,
                  float* C, int ldc, h16* Ch, const float* bias,
                  const float* add, int M, int N, int K) {
    dim3 g((N + 127) / 128, (M + 63) / 64);
    if (act == 0)      gemm_f16_wmma<0><<<g, 256, 0, stream>>>(A, lda, W, ldw, C, ldc, Ch, bias, add, M, N, K);
    else if (act == 1) gemm_f16_wmma<1><<<g, 256, 0, stream>>>(A, lda, W, ldw, C, ldc, Ch, bias, add, M, N, K);
    else               gemm_f16_wmma<2><<<g, 256, 0, stream>>>(A, lda, W, ldw, C, ldc, Ch, bias, add, M, N, K);
  };

  // weight conversion
  cvt(in_w,  in_w_h, D_ * 2 * DI_);
  cvt(xp_w,  xp_h,   DI_ * 32);
  cvt(dt_w,  dt_h,   DSTATE * DI_);
  cvt(out_w, outw_h, DI_ * D_);
  cvt(q_w,   qw_h,   D_ * D_);
  cvt(k_w,   kw_h,   D_ * 256);
  cvt(v_w,   vw_h,   D_ * 256);
  cvt(o_w,   ow_h,   D_ * D_);
  cvt(w1,    w1_h,   D_ * DFF);
  cvt(w2,    w2_h,   DFF * D_);
  cvt(w3,    w3_h,   D_ * DFF);

  // ---- Mamba block --------------------------------------------------------
  ln_kernel<<<T_, 256, 0, stream>>>(x_in, ln1_g, ln1_b, h1_h);
  gemm(0, h1_h, D_, in_w_h, 2 * DI_, xs_gate, 2 * DI_, nullptr, nullptr, nullptr, T_, 2 * DI_, D_);
  conv_silu_kernel<<<dim3((T_ * DI_) / 256), 256, 0, stream>>>(xs_gate, conv_w, conv_b, xsc, xsc_h);
  gemm(0, xsc_h, DI_, xp_h, 32, BC, 32, BC_h, nullptr, nullptr, T_, 32, DI_);
  gemm(2, BC_h, 32, dt_h, DI_, dtb, DI_, nullptr, dt_b, nullptr, T_, DI_, DSTATE);
  scan_kernel<<<dim3((B_ * DI_) / 256), 256, 0, stream>>>(xsc, dtb, BC, A_log, yb);
  gate_kernel<<<dim3((T_ * DI_) / 256), 256, 0, stream>>>(yb, xsc, xs_gate, Dskip, yg_h);
  gemm(0, yg_h, DI_, outw_h, D_, x1, D_, nullptr, nullptr, x_in, T_, D_, DI_);

  // ---- Attention block ----------------------------------------------------
  ln_kernel<<<T_, 256, 0, stream>>>(x1, ln2_g, ln2_b, h2_h);
  gemm(0, h2_h, D_, qw_h, D_, qf, D_, nullptr, nullptr, nullptr, T_, D_, D_);
  gemm(0, h2_h, D_, kw_h, 256, kf, 256, nullptr, nullptr, nullptr, T_, 256, D_);
  gemm(0, h2_h, D_, vw_h, 256, vf, 256, nullptr, nullptr, nullptr, T_, 256, D_);
  rope_q_kernel<<<dim3((B_ * NH * L_ * 32) / 256), 256, 0, stream>>>(qf, q_h);
  rope_k_kernel<<<dim3((B_ * NKV * L_ * 32) / 256), 256, 0, stream>>>(kf, k_t);
  vtrans_kernel<<<dim3((B_ * L_ * 256) / 256), 256, 0, stream>>>(vf, v_h);
  flash_attn<<<dim3(L_ / 16, NH, B_), 32, 0, stream>>>(q_h, k_t, v_h, o_h);
  gemm(0, o_h, D_, ow_h, D_, x2, D_, nullptr, nullptr, x1, T_, D_, D_);

  // ---- SwiGLU FFN ---------------------------------------------------------
  ln_kernel<<<T_, 256, 0, stream>>>(x2, ln3_g, ln3_b, h3_h);
  gemm(1, h3_h, D_, w1_h, DFF, u1, DFF, nullptr, nullptr, nullptr, T_, DFF, D_);
  gemm(0, h3_h, D_, w3_h, DFF, u3, DFF, nullptr, nullptr, nullptr, T_, DFF, D_);
  swiglu_kernel<<<dim3((T_ * DFF) / 256), 256, 0, stream>>>(u1, u3, g_h, T_ * DFF);
  gemm(0, g_h, DFF, w2_h, D_, out, D_, nullptr, nullptr, x2, T_, D_, DFF);
}
